// StandardSelfAttention_70712341561807
// MI455X (gfx1250) — compile-verified, matrix operand order checked by reading, not run on hardware
//
#include <hip/hip_runtime.h>
#include <hip/hip_bf16.h>

// ---------------------------------------------------------------------------
// Self-attention for MI455X (gfx1250, wave32, WMMA).
//   B=8, N=2048, D=1024, D_OUT=1024.
// Pipeline:
//   1) convert x / W_attn / W_out to f16 in workspace
//   2) WMMA GEMM: key16  = f16(x @ W_attn^T + b_attn)         [B*N,1024] row-major
//      WMMA GEMM: vT16   = f16(x @ W_out^T  + b_out), stored transposed [B,1024,2048]
//   3) flash attention: per 32 q-rows, stream 64-row K/V tiles, online softmax
//      (parallel across all 256 threads), S = Q K^T and O += P V via
//      v_wmma_f32_16x16x32_f16 with batched fragment preloads.
// ---------------------------------------------------------------------------

typedef _Float16 v8h  __attribute__((ext_vector_type(8)));
typedef _Float16 v16h __attribute__((ext_vector_type(16)));
typedef float    v8f  __attribute__((ext_vector_type(8)));

#define B_   8
#define N_   2048
#define D_   1024

// Build a 16-element f16 WMMA fragment from two contiguous 8-halves (2x b128).
// Matches the documented 16-bit A/B layout: lane holds one matrix row,
// elems [0..7] = K (h*8 .. h*8+7), elems [8..15] = K (16+h*8 .. 16+h*8+7).
static __device__ __forceinline__ v16h make_frag(const _Float16* lo,
                                                 const _Float16* hi) {
  v8h a = *(const v8h*)lo;
  v8h b = *(const v8h*)hi;
  return __builtin_shufflevector(a, b, 0, 1, 2, 3, 4, 5, 6, 7,
                                 8, 9, 10, 11, 12, 13, 14, 15);
}

static __device__ __forceinline__ v8f wmma_f16(v16h a, v16h b, v8f c) {
  return __builtin_amdgcn_wmma_f32_16x16x32_f16(false, a, false, b,
                                                (short)0, c, false, false);
}

// ---------------------------------------------------------------------------
// Kernel 0: f32 -> f16 conversion, 8 elems / thread.
// ---------------------------------------------------------------------------
__global__ __launch_bounds__(256) void cvt_f32_f16_kernel(
    const float* __restrict__ src, _Float16* __restrict__ dst, int n) {
  int i = (blockIdx.x * 256 + threadIdx.x) * 8;
  if (i + 8 <= n) {
    float4 a = *(const float4*)(src + i);
    float4 b = *(const float4*)(src + i + 4);
    v8h o = {(_Float16)a.x, (_Float16)a.y, (_Float16)a.z, (_Float16)a.w,
             (_Float16)b.x, (_Float16)b.y, (_Float16)b.z, (_Float16)b.w};
    *(v8h*)(dst + i) = o;
  }
}

// ---------------------------------------------------------------------------
// Kernel 1: Y = f16(X @ W^T + bias).  X:[M=16384,1024] f16, W:[1024,1024] f16.
// Block tile 128x128, 8 waves each compute 64x32 (4x2 subtiles of 16x16).
// K staged through LDS in 32-wide chunks (padded stride 40 f16).
// TRANSPOSE=1 stores Y[b][col][n] (for the later P @ V WMMA B-operand).
// ---------------------------------------------------------------------------
template <int TRANSPOSE>
__global__ __launch_bounds__(256) void gemm_xwT_kernel(
    const _Float16* __restrict__ X, const _Float16* __restrict__ W,
    const float* __restrict__ bias, _Float16* __restrict__ Y) {
  constexpr int KC = 32;
  constexpr int LDT = 40;  // padded LDS row stride (f16): conflict-free b128
  __shared__ _Float16 Xs[128 * LDT];
  __shared__ _Float16 Ws[128 * LDT];

  const int tid = threadIdx.x;
  const int lane = tid & 31;
  const int w = tid >> 5;
  const int l15 = lane & 15;
  const int h = lane >> 4;

  const int m0 = blockIdx.x * 128;
  const int n0 = blockIdx.y * 128;
  const int wm = (w >> 2) * 64;  // wave row base within tile
  const int wn = (w & 3) * 32;   // wave col base within tile

  v8f acc[4][2];
#pragma unroll
  for (int i = 0; i < 4; ++i)
#pragma unroll
    for (int j = 0; j < 2; ++j)
#pragma unroll
      for (int r = 0; r < 8; ++r) acc[i][j][r] = 0.f;

  const int srow = tid >> 1;        // 0..127
  const int scol = (tid & 1) * 16;  // 0 or 16

  for (int kb = 0; kb < D_; kb += KC) {
    // ---- stage X and W chunks into LDS (2x b128 each) ----
    {
      const _Float16* gx = X + (size_t)(m0 + srow) * D_ + kb + scol;
      v8h x0 = *(const v8h*)gx;
      v8h x1 = *(const v8h*)(gx + 8);
      *(v8h*)&Xs[srow * LDT + scol] = x0;
      *(v8h*)&Xs[srow * LDT + scol + 8] = x1;
      const _Float16* gw = W + (size_t)(n0 + srow) * D_ + kb + scol;
      v8h w0 = *(const v8h*)gw;
      v8h w1 = *(const v8h*)(gw + 8);
      *(v8h*)&Ws[srow * LDT + scol] = w0;
      *(v8h*)&Ws[srow * LDT + scol + 8] = w1;
    }
    __syncthreads();

    v16h bfrag[2];
#pragma unroll
    for (int ni = 0; ni < 2; ++ni) {
      const _Float16* p = &Ws[(wn + ni * 16 + l15) * LDT + h * 8];
      bfrag[ni] = make_frag(p, p + 16);
    }
#pragma unroll
    for (int mi = 0; mi < 4; ++mi) {
      const _Float16* p = &Xs[(wm + mi * 16 + l15) * LDT + h * 8];
      v16h afrag = make_frag(p, p + 16);
#pragma unroll
      for (int ni = 0; ni < 2; ++ni)
        acc[mi][ni] = wmma_f16(afrag, bfrag[ni], acc[mi][ni]);
    }
    __syncthreads();
  }

  // ---- epilogue: bias add, f16 convert, store ----
#pragma unroll
  for (int ni = 0; ni < 2; ++ni) {
    const int colg = n0 + wn + ni * 16 + l15;
    const float bv = bias[colg];
#pragma unroll
    for (int mi = 0; mi < 4; ++mi) {
      v8f c = acc[mi][ni];
#pragma unroll
      for (int r = 0; r < 8; ++r) {
        const int rowg = m0 + wm + mi * 16 + r + 8 * h;  // C layout: M=r+8h
        const _Float16 hv = (_Float16)(c[r] + bv);
        if (TRANSPOSE) {
          const int bb = rowg >> 11;      // row / 2048
          const int nn = rowg & 2047;     // row % 2048
          Y[((size_t)(bb * D_ + colg)) * N_ + nn] = hv;
        } else {
          Y[(size_t)rowg * D_ + colg] = hv;
        }
      }
    }
  }
}

// ---------------------------------------------------------------------------
// Kernel 2: flash attention.
// Block = (q-tile of 32 rows, batch). 8 waves:
//   Phase A: wave w computes S subtile (mi=w>>2, ni=w&3) of S[32][64]
//            over D=1024 in 32-wide WMMA chunks; scaled into LDS.
//   Phase B: online softmax, parallel over all 256 threads (8 threads/row,
//            shfl_xor butterflies inside aligned 8-lane groups; running m/l
//            replicated per-thread).
//   Phase C: every wave owns output cols [w*128, w*128+128): rescale its 16
//            accumulators by alpha, then O += P @ V via WMMA with all 8
//            V B-fragments preloaded per k-chunk (one load clause, then a
//            16-WMMA burst).
// ---------------------------------------------------------------------------
__global__ __launch_bounds__(256) void flash_attn_kernel(
    const _Float16* __restrict__ Q16,   // [B][N][D]
    const _Float16* __restrict__ K16,   // [B][N][D]
    const _Float16* __restrict__ VT16,  // [B][D][N]
    float* __restrict__ O) {            // [B][N][D]
  constexpr int TQ = 32, TK = 64;
  constexpr int SSTR = 68;  // f32 stride: 16B-aligned rows, low-conflict
  constexpr int PSTR = 72;  // f16 stride: conflict-free b128 frag loads
  __shared__ float Ss[TQ * SSTR];
  __shared__ _Float16 Ps[TQ * PSTR];
  __shared__ float alph[TQ];
  __shared__ float lbuf[TQ];

  const int tid = threadIdx.x;
  const int lane = tid & 31;
  const int w = tid >> 5;
  const int l15 = lane & 15;
  const int h = lane >> 4;
  const int q0 = blockIdx.x * TQ;
  const int bb = blockIdx.y;
  const int mi = w >> 2;   // S-phase row block 0..1
  const int ni = w & 3;    // S-phase col block 0..3
  const int o0 = w * 128;  // PV-phase: wave's output column base

  // softmax mapping: 8 threads per row, 8 elements per thread
  const int srow_id = tid >> 3;  // 0..31
  const int sseg = tid & 7;      // 0..7

  const float scale = 0.03125f;  // 1/sqrt(1024)

  v8f acc[2][8];
#pragma unroll
  for (int g = 0; g < 2; ++g)
#pragma unroll
    for (int oi = 0; oi < 8; ++oi)
#pragma unroll
      for (int r = 0; r < 8; ++r) acc[g][oi][r] = 0.f;

  float mrun = -__builtin_inff();  // per-thread replica for row srow_id
  float lrun = 0.f;

  const _Float16* qrow = Q16 + (size_t)(bb * N_ + q0 + mi * 16 + l15) * D_;
  const _Float16* kbat = K16 + (size_t)bb * N_ * D_;
  const _Float16* vbat = VT16 + (size_t)bb * D_ * N_;

  for (int k0 = 0; k0 < N_; k0 += TK) {
    // ---------------- Phase A: S = scale * Q K^T (one subtile per wave) ----
    v8f s;
#pragma unroll
    for (int r = 0; r < 8; ++r) s[r] = 0.f;
    const _Float16* krow = kbat + (size_t)(k0 + ni * 16 + l15) * D_;
    if (k0 + TK < N_) {  // hint the next K tile into cache
      __builtin_prefetch(kbat + (size_t)(k0 + TK + ni * 16 + l15) * D_ +
                             (size_t)h * 512,
                         0, 1);
    }
#pragma unroll 8
    for (int d = 0; d < D_; d += 32) {
      v16h aq = make_frag(qrow + d + h * 8, qrow + d + 16 + h * 8);
      v16h bk = make_frag(krow + d + h * 8, krow + d + 16 + h * 8);
      s = wmma_f16(aq, bk, s);
    }
#pragma unroll
    for (int r = 0; r < 8; ++r)
      Ss[(mi * 16 + r + 8 * h) * SSTR + ni * 16 + l15] = s[r] * scale;
    __syncthreads();

    // ---------------- Phase B: online softmax (all 256 threads) ------------
    {
      const float* srp = &Ss[srow_id * SSTR + sseg * 8];
      const float4 s0 = *(const float4*)(srp);
      const float4 s1 = *(const float4*)(srp + 4);
      float m8 = fmaxf(fmaxf(fmaxf(s0.x, s0.y), fmaxf(s0.z, s0.w)),
                       fmaxf(fmaxf(s1.x, s1.y), fmaxf(s1.z, s1.w)));
      // butterfly max across this row's 8 lanes (aligned group within wave)
      m8 = fmaxf(m8, __shfl_xor(m8, 1, 32));
      m8 = fmaxf(m8, __shfl_xor(m8, 2, 32));
      m8 = fmaxf(m8, __shfl_xor(m8, 4, 32));
      const float mnew = fmaxf(mrun, m8);
      const float al = __expf(mrun - mnew);
      const float p0 = __expf(s0.x - mnew), p1 = __expf(s0.y - mnew);
      const float p2 = __expf(s0.z - mnew), p3 = __expf(s0.w - mnew);
      const float p4 = __expf(s1.x - mnew), p5 = __expf(s1.y - mnew);
      const float p6 = __expf(s1.z - mnew), p7 = __expf(s1.w - mnew);
      v8h pk = {(_Float16)p0, (_Float16)p1, (_Float16)p2, (_Float16)p3,
                (_Float16)p4, (_Float16)p5, (_Float16)p6, (_Float16)p7};
      *(v8h*)&Ps[srow_id * PSTR + sseg * 8] = pk;
      float sum = ((p0 + p1) + (p2 + p3)) + ((p4 + p5) + (p6 + p7));
      sum += __shfl_xor(sum, 1, 32);
      sum += __shfl_xor(sum, 2, 32);
      sum += __shfl_xor(sum, 4, 32);
      lrun = lrun * al + sum;
      mrun = mnew;
      if (sseg == 0) alph[srow_id] = al;
    }
    __syncthreads();

    // ---------------- Phase C: rescale accumulators, O += P V -------------
#pragma unroll
    for (int g = 0; g < 2; ++g)
#pragma unroll
      for (int r = 0; r < 8; ++r) {
        const float av = alph[g * 16 + 8 * h + r];
#pragma unroll
        for (int oi = 0; oi < 8; ++oi) acc[g][oi][r] *= av;
      }
#pragma unroll
    for (int c = 0; c < 2; ++c) {
      v16h pf[2];
#pragma unroll
      for (int g = 0; g < 2; ++g) {
        const _Float16* pp = &Ps[(g * 16 + l15) * PSTR + c * 32 + h * 8];
        pf[g] = make_frag(pp, pp + 16);
      }
      // preload ALL eight V B-fragments -> one big load clause
      v16h bv[8];
#pragma unroll
      for (int oi = 0; oi < 8; ++oi) {
        const _Float16* vp =
            vbat + (size_t)(o0 + oi * 16 + l15) * N_ + k0 + c * 32 + h * 8;
        bv[oi] = make_frag(vp, vp + 16);
      }
      // 16-WMMA burst, no interleaved waits
#pragma unroll
      for (int oi = 0; oi < 8; ++oi)
#pragma unroll
        for (int g = 0; g < 2; ++g)
          acc[g][oi] = wmma_f16(pf[g], bv[oi], acc[g][oi]);
    }
    __syncthreads();  // protect Ss/Ps/alph for next tile
  }

  if (sseg == 0) lbuf[srow_id] = lrun;
  __syncthreads();

  // ---- final 1/l normalization + f32 store ----
#pragma unroll
  for (int g = 0; g < 2; ++g)
#pragma unroll
    for (int r = 0; r < 8; ++r) {
      const float inv = 1.f / lbuf[g * 16 + 8 * h + r];
      const int rowg = q0 + g * 16 + r + 8 * h;
      float* orow = O + (size_t)(bb * N_ + rowg) * D_;
#pragma unroll
      for (int oi = 0; oi < 8; ++oi)
        orow[o0 + oi * 16 + l15] = acc[g][oi][r] * inv;
    }
}

// ---------------------------------------------------------------------------
extern "C" void kernel_launch(void* const* d_in, const int* in_sizes, int n_in,
                              void* d_out, int out_size, void* d_ws,
                              size_t ws_size, hipStream_t stream) {
  const float* x = (const float*)d_in[0];
  const float* W_attn = (const float*)d_in[1];
  const float* b_attn = (const float*)d_in[2];
  const float* W_out = (const float*)d_in[3];
  const float* b_out = (const float*)d_in[4];
  float* out = (float*)d_out;

  const size_t XN = (size_t)B_ * N_ * D_;  // 16Mi elems
  const size_t WN = (size_t)D_ * D_;       // 1Mi elems
  char* ws = (char*)d_ws;
  _Float16* x16 = (_Float16*)ws;                         // 32 MiB
  _Float16* k16 = (_Float16*)(ws + XN * 2);              // 32 MiB
  _Float16* vT16 = (_Float16*)(ws + XN * 4);             // 32 MiB
  _Float16* wa16 = (_Float16*)(ws + XN * 6);             // 2 MiB
  _Float16* wo16 = (_Float16*)(ws + XN * 6 + WN * 2);    // 2 MiB

  // 1) f32 -> f16 conversions
  cvt_f32_f16_kernel<<<(int)(XN / 2048), 256, 0, stream>>>(x, x16, (int)XN);
  cvt_f32_f16_kernel<<<(int)(WN / 2048), 256, 0, stream>>>(W_attn, wa16,
                                                           (int)WN);
  cvt_f32_f16_kernel<<<(int)(WN / 2048), 256, 0, stream>>>(W_out, wo16,
                                                           (int)WN);

  // 2) projections: key (row-major) and value (transposed [B][D][N])
  gemm_xwT_kernel<0><<<dim3(128, 8), 256, 0, stream>>>(x16, wa16, b_attn, k16);
  gemm_xwT_kernel<1><<<dim3(128, 8), 256, 0, stream>>>(x16, wo16, b_out, vT16);

  // 3) fused attention
  flash_attn_kernel<<<dim3(N_ / 32, B_), 256, 0, stream>>>(x16, k16, vT16,
                                                           out);
}
